// Act24Weight16Matmul_19490561589769
// MI455X (gfx1250) — compile-verified, matrix operand order checked by reading, not run on hardware
//
#include <hip/hip_runtime.h>

// Split-f16 WMMA GEMM for Act24Weight16Matmul (x[4,8192,512] @ W[512,512]^T).
// x = x_hi(f16) + x_lo(f16), W = W_hi(f16) + W_lo(f16); three cross products
// (hi*hi, hi*lo, lo*hi) accumulated in f32 reproduce the reference's
// split-precision matmul to ~2^-21 relative — below the reference's GMAC
// output grid (2^-14 of global max), which also makes the lo*lo term (~2^-22)
// droppable, exactly as GMAC itself rounds y_lo_lo to ~0.

typedef __attribute__((ext_vector_type(16))) _Float16 v16h;
typedef __attribute__((ext_vector_type(8)))  _Float16 v8h;
typedef __attribute__((ext_vector_type(8)))  float    v8f;
typedef __attribute__((ext_vector_type(4)))  float    v4f;

#define KDIM 512
#define NDIM 512
#define NT   64     // N columns per block
#define KC   64     // K chunk staged in LDS per iteration (8 chunks total)
#define LSTR 72     // LDS row stride in halves (64 + 8 pad to spread banks)
#define WPB  8      // waves per block (256 threads, wave32)
#define MW   2      // 16-row A tiles per wave -> 32 M rows per wave

// ---- kernel 1: split weight into two f16 planes (exact: W == hi + lo to f16^2) ----
__global__ void prep_weight_split(const float* __restrict__ W,
                                  _Float16* __restrict__ WH,
                                  _Float16* __restrict__ WL) {
  int i = blockIdx.x * blockDim.x + threadIdx.x;   // grid sized exactly 512*512
  float w = W[i];
  _Float16 h = (_Float16)w;
  WH[i] = h;
  WL[i] = (_Float16)(w - (float)h);
}

// ---- kernel 2: tiled split-f16 GEMM, y[m,n] = sum_k x[m,k] * W[n,k] ----
__global__ __launch_bounds__(256) void split_f16_gemm(
    const float*    __restrict__ X,
    const _Float16* __restrict__ WH,
    const _Float16* __restrict__ WL,
    float*          __restrict__ Y) {
  __shared__ _Float16 sBH[NT][LSTR];
  __shared__ _Float16 sBL[NT][LSTR];

  const int tid    = threadIdx.x;
  const int wave   = tid >> 5;          // 0..7
  const int lane   = tid & 31;          // wave32
  const int lane16 = lane & 15;
  const int khalf  = (lane >> 4) << 3;  // 0 or 8: K sub-offset per half-wave (ISA 7.12.2)

  const int n0 = blockIdx.x * NT;
  const int m0 = (blockIdx.y * WPB + wave) * (16 * MW);

  long arow[MW];
#pragma unroll
  for (int mi = 0; mi < MW; ++mi)
    arow[mi] = (long)(m0 + mi * 16 + lane16) * KDIM;   // A: lane%16 -> M row

  // staging indices for cooperative W tile loads (16 halves/thread/plane)
  const int sr = tid >> 2;              // 0..63 : n row in tile
  const int sc = (tid & 3) << 4;        // 0,16,32,48 : k offset
  const _Float16* gH = WH + (long)(n0 + sr) * KDIM + sc;
  const _Float16* gL = WL + (long)(n0 + sr) * KDIM + sc;

  v8f acc[MW][4] = {};                  // MW x 4 tiles of 16x16 f32

  // ---- software pipeline: hold next K-chunk of W planes in registers ----
  v8h rh0 = *(const v8h*)(gH);
  v8h rh1 = *(const v8h*)(gH + 8);
  v8h rl0 = *(const v8h*)(gL);
  v8h rl1 = *(const v8h*)(gL + 8);

  for (int kc = 0; kc < KDIM; kc += KC) {
    __syncthreads();                    // previous chunk's compute done
    *(v8h*)&sBH[sr][sc]     = rh0;
    *(v8h*)&sBH[sr][sc + 8] = rh1;
    *(v8h*)&sBL[sr][sc]     = rl0;
    *(v8h*)&sBL[sr][sc + 8] = rl1;
    __syncthreads();                    // staged chunk visible

    if (kc + KC < KDIM) {               // prefetch next chunk: overlaps compute
      rh0 = *(const v8h*)(gH + kc + KC);
      rh1 = *(const v8h*)(gH + kc + KC + 8);
      rl0 = *(const v8h*)(gL + kc + KC);
      rl1 = *(const v8h*)(gL + kc + KC + 8);
      __builtin_prefetch(X + arow[0] + kc + KC, 0, 1);   // global_prefetch_b8
    }

#pragma unroll
    for (int ks = 0; ks < KC; ks += 32) {
      // --- A fragments: load 16 f32 of x per tile, split into hi/lo f16.
      // v16h layout: elems 0..7 -> K = base+khalf+0..7, elems 8..15 -> +16..23.
      v16h ahi[MW], alo[MW];
#pragma unroll
      for (int mi = 0; mi < MW; ++mi) {
        const float* xp = X + arow[mi] + kc + ks + khalf;
        union { v4f v[4]; float f[16]; } ax;
        ax.v[0] = *(const v4f*)(xp);
        ax.v[1] = *(const v4f*)(xp + 4);
        ax.v[2] = *(const v4f*)(xp + 16);
        ax.v[3] = *(const v4f*)(xp + 20);
#pragma unroll
        for (int j = 0; j < 16; ++j) {
          float    f = ax.f[j];
          _Float16 h = (_Float16)f;
          ahi[mi][j] = h;
          alo[mi][j] = (_Float16)(f - (float)h);
        }
      }

      const int cc = ks + khalf;
#pragma unroll
      for (int t = 0; t < 4; ++t) {
        // --- B fragment from LDS: lane%16 -> N column, same K split as A.
        const int nr = (t << 4) + lane16;
        v8h h0 = *(const v8h*)&sBH[nr][cc];
        v8h h1 = *(const v8h*)&sBH[nr][cc + 16];
        v8h l0 = *(const v8h*)&sBL[nr][cc];
        v8h l1 = *(const v8h*)&sBL[nr][cc + 16];
        v16h bhi = __builtin_shufflevector(h0, h1, 0,1,2,3,4,5,6,7,8,9,10,11,12,13,14,15);
        v16h blo = __builtin_shufflevector(l0, l1, 0,1,2,3,4,5,6,7,8,9,10,11,12,13,14,15);

#pragma unroll
        for (int mi = 0; mi < MW; ++mi) {
          acc[mi][t] = __builtin_amdgcn_wmma_f32_16x16x32_f16(false, ahi[mi], false, bhi,
                                                              (short)0, acc[mi][t], false, false);
          acc[mi][t] = __builtin_amdgcn_wmma_f32_16x16x32_f16(false, ahi[mi], false, blo,
                                                              (short)0, acc[mi][t], false, false);
          acc[mi][t] = __builtin_amdgcn_wmma_f32_16x16x32_f16(false, alo[mi], false, bhi,
                                                              (short)0, acc[mi][t], false, false);
          // lo*lo omitted: ~2^-22 relative, below the reference GMAC grid
        }
      }
    }
  }

  // C/D layout (ISA 7.12.2): lane%16 -> N col; VGPR r -> M row r (+8 for lanes 16..31)
#pragma unroll
  for (int mi = 0; mi < MW; ++mi) {
    const int rbase = m0 + mi * 16 + ((lane >> 4) << 3);
#pragma unroll
    for (int t = 0; t < 4; ++t) {
      float* yp = Y + (long)rbase * NDIM + n0 + (t << 4) + lane16;
#pragma unroll
      for (int r = 0; r < 8; ++r)
        yp[(long)r * NDIM] = acc[mi][t][r];
    }
  }
}

extern "C" void kernel_launch(void* const* d_in, const int* in_sizes, int n_in,
                              void* d_out, int out_size, void* d_ws, size_t ws_size,
                              hipStream_t stream) {
  (void)n_in; (void)out_size; (void)ws_size;
  const float* X = (const float*)d_in[0];   // [4,8192,512] f32
  const float* W = (const float*)d_in[1];   // [512,512]    f32
  float*       Y = (float*)d_out;           // [4,8192,512] f32

  const int M = in_sizes[0] / KDIM;         // 32768 (multiple of 256)

  _Float16* WH = (_Float16*)d_ws;                       // 512*512 halves
  _Float16* WL = WH + (size_t)NDIM * KDIM;              // 512*512 halves (1 MB total)

  prep_weight_split<<<(NDIM * KDIM) / 256, 256, 0, stream>>>(W, WH, WL);

  dim3 grid(NDIM / NT, M / (WPB * 16 * MW));            // (8, 128)
  split_f16_gemm<<<grid, 256, 0, stream>>>(X, WH, WL, Y);
}
